// DynamicShortConvolution_51651276701955
// MI455X (gfx1250) — compile-verified, matrix operand order checked by reading, not run on hardware
//
#include <hip/hip_runtime.h>
#include <hip/hip_bf16.h>

typedef __attribute__((ext_vector_type(16))) _Float16 v16h;
typedef __attribute__((ext_vector_type(8)))  _Float16 v8h;
typedef __attribute__((ext_vector_type(8)))  float    v8f;
typedef __attribute__((ext_vector_type(4)))  float    v4f;

// Problem constants (from reference setup_inputs)
#define BB 4
#define TT 4096
#define DD 2048
#define HH 256
#define WW 4
#define OO (DD * WW)          // 8192
#define MM (BB * TT)          // 16384 tokens

// LDS strip layout: 64 rows x 256 halves, padded to 264 halves
// (528 B = 33*16 B stride keeps b128 lane accesses on distinct bank groups)
#define BROW_PAD 264

__device__ __forceinline__ float silu_f(float v) {
    // v * rcp(1+exp(-v)): v_exp_f32 + v_rcp_f32, no IEEE-div fixup chain
    return v * __builtin_amdgcn_rcpf(1.0f + __expf(-v));
}

// ---------------------------------------------------------------------------
// Weight transpose + fp32->fp16 convert.
// w1 [D,H] row-major -> w1t [H,D] f16 ;  w2 [H,O] row-major -> w2t [O,H] f16
// ---------------------------------------------------------------------------
__global__ void cvt_w1_kernel(const float* __restrict__ w1, _Float16* __restrict__ w1t) {
    int idx = blockIdx.x * blockDim.x + threadIdx.x;      // over D*H, h fastest
    int d = idx / HH;
    int h = idx % HH;
    w1t[(size_t)h * DD + d] = (_Float16)w1[idx];
}

__global__ void cvt_w2_kernel(const float* __restrict__ w2, _Float16* __restrict__ w2t) {
    int idx = blockIdx.x * blockDim.x + threadIdx.x;      // over H*O, o fastest
    int h = idx / OO;
    int o = idx % OO;
    w2t[(size_t)o * HH + h] = (_Float16)w2[idx];
}

// ---------------------------------------------------------------------------
// GEMM1: h = silu(x @ w1), stored f16 row-major [MM, HH].
// Block = 8 waves sharing one 64-col w1t strip; K = 2048 is staged through
// LDS in eight 32 KB chunks, double-buffered: chunk k+1 is loaded while
// computing on chunk k, so weight-load latency hides behind 32 WMMAs/chunk.
// Each wave owns a 16-token M-tile, converts x fp32->f16 once per 4 tiles.
// ---------------------------------------------------------------------------
__global__ void gemm1_silu_kernel(const float* __restrict__ x,
                                  const _Float16* __restrict__ w1t,
                                  _Float16* __restrict__ hout) {
    __shared__ _Float16 bsh[2][64 * BROW_PAD];      // 2 x 33 KB

    const int ngroups = HH / 64;                    // 4
    const int ngrp = blockIdx.x % ngroups;          // strips adjacent -> x L2 reuse
    const int mblk = blockIdx.x / ngroups;          // 128 m-blocks
    const int wid  = threadIdx.x >> 5;
    const int lane = threadIdx.x & 31;
    const int ln = lane & 15;
    const int kg = lane >> 4;

    const int tile_m = mblk * 8 + wid;
    const float* arow = x + (size_t)(tile_m * 16 + ln) * DD;
    const char* bb0 = (const char*)&bsh[0][0] + (size_t)ln * (BROW_PAD * 2);
    const char* bb1 = (const char*)&bsh[1][0] + (size_t)ln * (BROW_PAD * 2);

    auto load_chunk = [&](int buf, int kc) {
#pragma unroll
        for (int i = 0; i < 8; ++i) {
            const int c   = threadIdx.x + i * 256;  // 16 B chunk id, 2048 total
            const int row = c >> 5;                 // 32 chunks per 512 B row
            const int col = c & 31;
            *(v8h*)((char*)&bsh[buf][0] + (size_t)row * (BROW_PAD * 2) + col * 16) =
                *(const v8h*)(w1t + (size_t)(ngrp * 64 + row) * DD + kc * 256 + col * 8);
        }
    };

    v8f acc0 = {}, acc1 = {}, acc2 = {}, acc3 = {};
    load_chunk(0, 0);
    __syncthreads();
    for (int kc = 0; kc < 8; ++kc) {
        if (kc < 7) load_chunk((kc + 1) & 1, kc + 1);   // pipeline next chunk
        const char* bb = (kc & 1) ? bb1 : bb0;
        const int kbase = kc * 256;
#pragma unroll
        for (int kk = 0; kk < 256; kk += 32) {
            const int ka = kk + kg * 8;
            const int kb = kk + 16 + kg * 8;
            __builtin_prefetch(arow + kbase + kk + 64, 0, 1);  // global_prefetch_b8
            v4f a0 = *(const v4f*)(arow + kbase + ka);
            v4f a1 = *(const v4f*)(arow + kbase + ka + 4);
            v4f a2 = *(const v4f*)(arow + kbase + kb);
            v4f a3 = *(const v4f*)(arow + kbase + kb + 4);
            v16h afrag;
#pragma unroll
            for (int j = 0; j < 4; ++j) {
                afrag[j]      = (_Float16)a0[j];
                afrag[4 + j]  = (_Float16)a1[j];
                afrag[8 + j]  = (_Float16)a2[j];
                afrag[12 + j] = (_Float16)a3[j];
            }
            union { v16h v; v8h h2[2]; } b0, b1, b2s, b3;
            b0.h2[0]  = *(const v8h*)(bb + (size_t)( 0) * (BROW_PAD * 2) + ka * 2);
            b0.h2[1]  = *(const v8h*)(bb + (size_t)( 0) * (BROW_PAD * 2) + kb * 2);
            b1.h2[0]  = *(const v8h*)(bb + (size_t)(16) * (BROW_PAD * 2) + ka * 2);
            b1.h2[1]  = *(const v8h*)(bb + (size_t)(16) * (BROW_PAD * 2) + kb * 2);
            b2s.h2[0] = *(const v8h*)(bb + (size_t)(32) * (BROW_PAD * 2) + ka * 2);
            b2s.h2[1] = *(const v8h*)(bb + (size_t)(32) * (BROW_PAD * 2) + kb * 2);
            b3.h2[0]  = *(const v8h*)(bb + (size_t)(48) * (BROW_PAD * 2) + ka * 2);
            b3.h2[1]  = *(const v8h*)(bb + (size_t)(48) * (BROW_PAD * 2) + kb * 2);
            acc0 = __builtin_amdgcn_wmma_f32_16x16x32_f16(false, afrag, false, b0.v,  (short)0, acc0, false, false);
            acc1 = __builtin_amdgcn_wmma_f32_16x16x32_f16(false, afrag, false, b1.v,  (short)0, acc1, false, false);
            acc2 = __builtin_amdgcn_wmma_f32_16x16x32_f16(false, afrag, false, b2s.v, (short)0, acc2, false, false);
            acc3 = __builtin_amdgcn_wmma_f32_16x16x32_f16(false, afrag, false, b3.v,  (short)0, acc3, false, false);
        }
        __syncthreads();
    }

    const int rowbase = tile_m * 16 + kg * 8;
    const int colbase = ngrp * 64 + ln;
    v8f* accs[4] = {&acc0, &acc1, &acc2, &acc3};
#pragma unroll
    for (int s = 0; s < 4; ++s) {
        v8f a = *accs[s];
#pragma unroll
        for (int r = 0; r < 8; ++r) {
            hout[(size_t)(rowbase + r) * HH + (colbase + s * 16)] = (_Float16)silu_f(a[r]);
        }
    }
}

// ---------------------------------------------------------------------------
// GEMM2 fused with bias + causal dynamic depthwise conv + silu.
// Block (8 waves) shares one 64-col w2t strip staged in LDS (32 KB, padded);
// each wave handles its own 16-token M-tile -> 128 tokens/block.
// K = HH = 256 -> 8 unrolled K-steps x 4 WMMAs = 32 WMMAs/wave, with B
// fragments served by ds_load_b128 instead of per-WMMA global round-trips.
// o = 4*d + w  =>  each 16-col tile covers 4 channels x 4 taps; lane's flat
// value (+bias) is multiplied by x[b, t-3+w, d], the 4 tap-lanes are summed
// via ds_swizzle xor-reduce, lead lane applies silu and stores out[b,t,d].
// `flat` (512 MiB) never touches HBM.
// ---------------------------------------------------------------------------
__global__ void gemm2_conv_silu_kernel(const _Float16* __restrict__ hin,
                                       const _Float16* __restrict__ w2t,
                                       const float* __restrict__ b2,
                                       const float* __restrict__ x,
                                       float* __restrict__ out) {
    __shared__ _Float16 bsh[64 * BROW_PAD];        // 33 KB

    const int ngroups = OO / 64;                   // 128
    const int ngrp = blockIdx.x % ngroups;
    const int mblk = blockIdx.x / ngroups;         // 128 blocks of 8 m-tiles
    const int wid  = threadIdx.x >> 5;
    const int lane = threadIdx.x & 31;
    const int ln = lane & 15;
    const int kg = lane >> 4;

    // Cooperative load of the contiguous 32 KB strip w2t[ngrp*64 .. +64) rows.
    {
        const _Float16* bstrip = w2t + (size_t)ngrp * 64 * HH;
#pragma unroll
        for (int i = 0; i < 8; ++i) {
            const int c   = threadIdx.x + i * 256;  // 16-byte chunk id, 2048 total
            const int row = c >> 5;                 // 32 chunks per 512 B row
            const int col = c & 31;
            *(v8h*)((char*)bsh + (size_t)row * (BROW_PAD * 2) + col * 16) =
                *(const v8h*)((const char*)bstrip + (size_t)c * 16);
        }
    }
    __syncthreads();

    const int tile_m = mblk * 8 + wid;             // this wave's 16-token tile
    const _Float16* arow = hin + (size_t)(tile_m * 16 + ln) * HH;
    const char* bbase = (const char*)bsh + (size_t)ln * (BROW_PAD * 2);

    v8f acc0 = {}, acc1 = {}, acc2 = {}, acc3 = {};
#pragma unroll
    for (int k0 = 0; k0 < HH; k0 += 32) {
        const int ka = k0 + kg * 8;
        const int kb = k0 + 16 + kg * 8;
        union { v16h v; v8h h2[2]; } au, b0, b1, b2s, b3;
        au.h2[0] = *(const v8h*)(arow + ka);
        au.h2[1] = *(const v8h*)(arow + kb);
        b0.h2[0]  = *(const v8h*)(bbase + (size_t)( 0) * (BROW_PAD * 2) + ka * 2);
        b0.h2[1]  = *(const v8h*)(bbase + (size_t)( 0) * (BROW_PAD * 2) + kb * 2);
        b1.h2[0]  = *(const v8h*)(bbase + (size_t)(16) * (BROW_PAD * 2) + ka * 2);
        b1.h2[1]  = *(const v8h*)(bbase + (size_t)(16) * (BROW_PAD * 2) + kb * 2);
        b2s.h2[0] = *(const v8h*)(bbase + (size_t)(32) * (BROW_PAD * 2) + ka * 2);
        b2s.h2[1] = *(const v8h*)(bbase + (size_t)(32) * (BROW_PAD * 2) + kb * 2);
        b3.h2[0]  = *(const v8h*)(bbase + (size_t)(48) * (BROW_PAD * 2) + ka * 2);
        b3.h2[1]  = *(const v8h*)(bbase + (size_t)(48) * (BROW_PAD * 2) + kb * 2);
        acc0 = __builtin_amdgcn_wmma_f32_16x16x32_f16(false, au.v, false, b0.v,  (short)0, acc0, false, false);
        acc1 = __builtin_amdgcn_wmma_f32_16x16x32_f16(false, au.v, false, b1.v,  (short)0, acc1, false, false);
        acc2 = __builtin_amdgcn_wmma_f32_16x16x32_f16(false, au.v, false, b2s.v, (short)0, acc2, false, false);
        acc3 = __builtin_amdgcn_wmma_f32_16x16x32_f16(false, au.v, false, b3.v,  (short)0, acc3, false, false);
    }

    const int m0 = tile_m * 16 + kg * 8;           // first token row in lane
    const int b  = m0 >> 12;                       // m / TT (tiles never cross batch)

    v8f* accs[4] = {&acc0, &acc1, &acc2, &acc3};
#pragma unroll
    for (int s = 0; s < 4; ++s) {
        const int o = ngrp * 64 + s * 16 + ln;     // flat column
        const float bias = b2[o];
        const int d = o >> 2;                      // channel
        const int w = o & 3;                       // tap index
        v8f a = *accs[s];
        float res[8];
#pragma unroll
        for (int r = 0; r < 8; ++r) {
            const int m  = m0 + r;
            const int t  = m & (TT - 1);
            const int ts = t - (WW - 1) + w;       // shifted source time (causal pad)
            const float xv = (ts >= 0)
                ? x[((size_t)b * TT + ts) * DD + d] : 0.0f;
            float p = (a[r] + bias) * xv;
            p += __builtin_bit_cast(float,
                  __builtin_amdgcn_ds_swizzle(__builtin_bit_cast(int, p), 0x041F)); // xor 1
            p += __builtin_bit_cast(float,
                  __builtin_amdgcn_ds_swizzle(__builtin_bit_cast(int, p), 0x081F)); // xor 2
            res[r] = p;
        }
        if (w == 0) {
#pragma unroll
            for (int r = 0; r < 8; ++r) {
                const int m = m0 + r;
                const int t = m & (TT - 1);
                out[((size_t)b * TT + t) * DD + d] = silu_f(res[r]);
            }
        }
    }
}

// ---------------------------------------------------------------------------
// Launch: inputs are {x, w1, w2, b2} fp32; output fp32 [B,T,D].
// Workspace: w1t f16 (1 MiB) | w2t f16 (4 MiB) | h f16 (8 MiB) = 13 MiB.
// ---------------------------------------------------------------------------
extern "C" void kernel_launch(void* const* d_in, const int* in_sizes, int n_in,
                              void* d_out, int out_size, void* d_ws, size_t ws_size,
                              hipStream_t stream) {
    (void)in_sizes; (void)n_in; (void)out_size; (void)ws_size;
    const float* x  = (const float*)d_in[0];
    const float* w1 = (const float*)d_in[1];
    const float* w2 = (const float*)d_in[2];
    const float* b2 = (const float*)d_in[3];
    float* out = (float*)d_out;

    char* ws = (char*)d_ws;
    _Float16* w1t  = (_Float16*)ws;                                  // DD*HH halves
    _Float16* w2t  = (_Float16*)(ws + (size_t)DD * HH * 2);          // OO*HH halves
    _Float16* hbuf = (_Float16*)(ws + (size_t)DD * HH * 2
                                    + (size_t)OO * HH * 2);          // MM*HH halves

    // Weight convert/transpose
    cvt_w1_kernel<<<(DD * HH) / 256, 256, 0, stream>>>(w1, w1t);
    cvt_w2_kernel<<<(HH * OO) / 256, 256, 0, stream>>>(w2, w2t);

    // GEMM1: 128 m-blocks x 4 strips = 512 blocks, 8 waves each
    {
        const int blocks = (MM / 128) * (HH / 64);
        gemm1_silu_kernel<<<blocks, 256, 0, stream>>>(x, w1t, hbuf);
    }
    // GEMM2 fused epilogue: 128 m-blocks x 128 strips = 16384 blocks
    {
        const int blocks = (MM / 128) * (OO / 64);
        gemm2_conv_silu_kernel<<<blocks, 256, 0, stream>>>(hbuf, w2t, b2, x, out);
    }
}
